// UEDDIENetwork_33028298506446
// MI455X (gfx1250) — matrix operand order repaired and checked by reading, not compiled, not run on hardware
//
#include <hip/hip_runtime.h>
#include <hip/hip_bf16.h>
#include <math.h>

// ---------------------------------------------------------------------------
// Funnel dims: [512,128,115,102,90,77,64,51,38,26,13,1]  (11 linear layers)
// K padded to mult of 32 (WMMA K-step), N padded to mult of 16 (WMMA N-tile).
// ---------------------------------------------------------------------------
typedef __attribute__((ext_vector_type(16))) _Float16 v16h;
typedef __attribute__((ext_vector_type(8)))  float    v8f;
typedef __attribute__((ext_vector_type(4)))  float    f32x4;

constexpr int NLAYER = 11;
constexpr int L_IN [NLAYER] = {512,128,115,102, 90, 77, 64, 51, 38, 26, 13};
constexpr int L_OUT[NLAYER] = {128,115,102, 90, 77, 64, 51, 38, 26, 13,  1};
constexpr int L_KS [NLAYER] = { 16,  4,  4,  4,  3,  3,  2,  2,  2,  1,  1}; // Kpad/32
constexpr int L_NT [NLAYER] = {  8,  8,  7,  6,  5,  4,  4,  3,  2,  1,  1}; // Npad/16
// Offsets (in halves) of each layer's packed fragments within one expert's block
constexpr int WOFF[NLAYER] = {0,65536,81920,96256,108544,116224,122368,126464,129536,131584,132096};
constexpr int WSTRIDE = 132608;                 // halves per expert
constexpr int BOFF[NLAYER] = {0,128,256,368,464,544,608,672,720,752,768};
constexpr int BSTRIDE = 784;                    // padded bias floats per expert

constexpr int NEXPERT = 7;                      // 4 element + 3 charge
constexpr int NATOM   = 1024 * 64;              // B*A
constexpr int TILE_M  = 32;                     // atoms per workgroup (2 WMMA row-blocks)
constexpr int NTILES  = NATOM / TILE_M;         // 2048

constexpr size_t WBUF_BYTES = (size_t)NEXPERT * WSTRIDE * 2;   // 1,856,512
constexpr size_t BBUF_BYTES = (size_t)NEXPERT * BSTRIDE * 4;   //    21,952
// y table: [7][65536] f32 follows in workspace

struct WPtrs {                       // [0..10] = elem group, [11..21] = chg group
    const float* W[22];
    const float* Bv[22];
};

// fast softplus: exp/log map to TRANS32 units that co-execute with WMMA
__device__ __forceinline__ float softplus_fast(float x) {
    return (x > 15.0f) ? x : __logf(1.0f + __expf(x));
}

// ---------------------------------------------------------------------------
// Kernel 1: pack weights into WMMA B-fragment layout (+ padded biases).
// B layout for 16-bit 32x16 (per ISA sparse-B doc): lane n = L%16, half = L/16,
// half-slot hh (0..15) holds K = (hh&7) + 8*half + 16*(hh>>3). Zero-fill K>=in, N>=out so the
// padded regions contribute exactly zero in the GEMM.
// ---------------------------------------------------------------------------
__global__ __launch_bounds__(256) void prep_kernel(WPtrs p, _Float16* wbuf, float* bbuf)
{
    const int e = blockIdx.x / NLAYER;           // expert 0..6
    const int l = blockIdx.x % NLAYER;           // layer  0..10
    const int grp = (e < 4) ? 0 : 1;
    const int el  = (e < 4) ? e : e - 4;
    const float* Wsrc = p.W [grp * NLAYER + l];
    const float* Bsrc = p.Bv[grp * NLAYER + l];
    const int in  = L_IN[l],  out = L_OUT[l];
    const int KS  = L_KS[l],  NT  = L_NT[l];

    _Float16* wd = wbuf + (size_t)e * WSTRIDE + WOFF[l];
    const int total = KS * NT * 512;             // halves this (e,l)
    for (int i = threadIdx.x; i < total; i += blockDim.x) {
        const int frag = i >> 9;                 // which 32x16 fragment
        const int r    = i & 511;
        const int lane = r >> 4;
        const int hh   = r & 15;
        const int ks   = frag / NT;
        const int nt   = frag % NT;
        const int half = lane >> 4;
        const int n    = nt * 16 + (lane & 15);
        const int k    = ks * 32 + (hh & 7) + 8 * half + 16 * (hh >> 3);
        float v = (n < out && k < in)
                ? Wsrc[(size_t)el * out * in + (size_t)n * in + k] : 0.0f;
        wd[i] = (_Float16)v;
    }
    float* bd = bbuf + e * BSTRIDE + BOFF[l];
    const int npad = NT * 16;
    for (int i = threadIdx.x; i < npad; i += blockDim.x)
        bd[i] = (i < out) ? Bsrc[el * out + i] : 0.0f;
}

// ---------------------------------------------------------------------------
// One funnel layer for a 32-atom tile, one wave, fully unrolled.
//   Two 16-row A blocks share every B fragment (2x B reuse vs M=16).
//   A (16xKpad f16) from LDS: 16-bit A layout => per lane two contiguous
//   K-octets -> two 16B ds loads per k-step per block.
//   B from packed global fragments: 32B contiguous per lane.
//   C tile layout: VGPR r, lane -> (m = r + 8*(lane/16), n = nt*16 + lane%16).
// ---------------------------------------------------------------------------
template<int KS, int NT, int SWIN, bool FINAL>
__device__ __forceinline__
void funnel_layer(const _Float16* inLDS,        // input activations (LDS)
                  _Float16* outLDS,             // output activations (LDS, stride 128)
                  const _Float16* __restrict__ wfrag,  // packed weights (global)
                  const float*    __restrict__ bias,   // padded bias (global)
                  float* yout,                  // final per-atom outputs (FINAL only)
                  int lane)
{
    const int m   = lane & 15;
    const int hlf = lane >> 4;

    __builtin_prefetch(wfrag + (size_t)lane * 64, 0, 0);   // warm the fragment stream

    v8f acc0[NT] = {};                          // rows 0..15
    v8f acc1[NT] = {};                          // rows 16..31
#pragma unroll
    for (int ks = 0; ks < KS; ++ks) {
        union { f32x4 q[2]; v16h h; } A0, A1;
        const int k0 = ks * 32 + hlf * 8;       // this lane's first K-octet
        A0.q[0] = *(const f32x4*)(inLDS + m * SWIN + k0);
        A0.q[1] = *(const f32x4*)(inLDS + m * SWIN + k0 + 16);
        A1.q[0] = *(const f32x4*)(inLDS + (16 + m) * SWIN + k0);
        A1.q[1] = *(const f32x4*)(inLDS + (16 + m) * SWIN + k0 + 16);
#pragma unroll
        for (int nt = 0; nt < NT; ++nt) {
            union { f32x4 q[2]; v16h h; } Bf;
            const _Float16* wp = wfrag + (ks * NT + nt) * 512 + lane * 16;
            Bf.q[0] = *(const f32x4*)(wp);
            Bf.q[1] = *(const f32x4*)(wp + 8);
            acc0[nt] = __builtin_amdgcn_wmma_f32_16x16x32_f16(
                false, A0.h, false, Bf.h, (short)0, acc0[nt], false, false);
            acc1[nt] = __builtin_amdgcn_wmma_f32_16x16x32_f16(
                false, A1.h, false, Bf.h, (short)0, acc1[nt], false, false);
        }
    }

#pragma unroll
    for (int nt = 0; nt < NT; ++nt) {
        const float bval = bias[nt * 16 + m];
        if constexpr (!FINAL) {
#pragma unroll
            for (int r = 0; r < 8; ++r) {
                const int row = r + hlf * 8;
                const int col = nt * 16 + m;
                outLDS[row * 128 + col]        = (_Float16)softplus_fast(acc0[nt][r] + bval);
                outLDS[(16 + row) * 128 + col] = (_Float16)softplus_fast(acc1[nt][r] + bval);
            }
        } else {
            if (m == 0) {                        // column 0 carries the scalar out
#pragma unroll
                for (int r = 0; r < 8; ++r) {
                    yout[hlf * 8 + r]      = acc0[nt][r] + bval;
                    yout[16 + hlf * 8 + r] = acc1[nt][r] + bval;
                }
            }
        }
    }
}

// ---------------------------------------------------------------------------
// Kernel 2: fused 7-expert funnel. One workgroup = 32 atoms, 7 waves = 7
// experts sharing the X tile in LDS (X read once from HBM).
// ---------------------------------------------------------------------------
__global__ __launch_bounds__(224) void funnel_kernel(const float* __restrict__ X,
                                                     const _Float16* __restrict__ wbuf,
                                                     const float* __restrict__ bbuf,
                                                     float* __restrict__ y)
{
    extern __shared__ __align__(16) char smem[];
    _Float16* xtile = (_Float16*)smem;                  // [32][512] f16 = 32 KB
    _Float16* htile = xtile + TILE_M * 512;             // 7 x [32][128] f16 = 56 KB

    const int tid  = threadIdx.x;
    const int wave = tid >> 5;                          // expert id 0..6
    const int lane = tid & 31;
    const int tile = blockIdx.x;                        // atom tile 0..2047

    // zero activation tiles once (keeps padded columns finite/zero for WMMA)
    for (int i = tid; i < NEXPERT * TILE_M * 128 / 2; i += 224)
        ((float*)htile)[i] = 0.0f;

    // stage X tile f32 -> f16
    const float* xg = X + (size_t)tile * TILE_M * 512;
    for (int i = tid; i < TILE_M * 512; i += 224)
        xtile[i] = (_Float16)xg[i];
    __syncthreads();

    const _Float16* wbase = wbuf + (size_t)wave * WSTRIDE;
    const float*    bbase = bbuf + wave * BSTRIDE;
    _Float16* h    = htile + wave * TILE_M * 128;
    float*    yout = y + (size_t)wave * NATOM + tile * TILE_M;

    funnel_layer<16, 8, 512, false>(xtile, h, wbase + WOFF[0],  bbase + BOFF[0],  nullptr, lane);
    funnel_layer< 4, 8, 128, false>(h,     h, wbase + WOFF[1],  bbase + BOFF[1],  nullptr, lane);
    funnel_layer< 4, 7, 128, false>(h,     h, wbase + WOFF[2],  bbase + BOFF[2],  nullptr, lane);
    funnel_layer< 4, 6, 128, false>(h,     h, wbase + WOFF[3],  bbase + BOFF[3],  nullptr, lane);
    funnel_layer< 3, 5, 128, false>(h,     h, wbase + WOFF[4],  bbase + BOFF[4],  nullptr, lane);
    funnel_layer< 3, 4, 128, false>(h,     h, wbase + WOFF[5],  bbase + BOFF[5],  nullptr, lane);
    funnel_layer< 2, 4, 128, false>(h,     h, wbase + WOFF[6],  bbase + BOFF[6],  nullptr, lane);
    funnel_layer< 2, 3, 128, false>(h,     h, wbase + WOFF[7],  bbase + BOFF[7],  nullptr, lane);
    funnel_layer< 2, 2, 128, false>(h,     h, wbase + WOFF[8],  bbase + BOFF[8],  nullptr, lane);
    funnel_layer< 1, 1, 128, false>(h,     h, wbase + WOFF[9],  bbase + BOFF[9],  nullptr, lane);
    funnel_layer< 1, 1, 128, true >(h,     h, wbase + WOFF[10], bbase + BOFF[10], yout,    lane);
}

// ---------------------------------------------------------------------------
// Kernel 3: route by E / C+1, reduce over A=64 per molecule, out = -ΣIE*ΣCF.
// One wave per molecule b; 8 waves per block.
// ---------------------------------------------------------------------------
__global__ __launch_bounds__(256) void combine_kernel(const int* __restrict__ E,
                                                      const int* __restrict__ C,
                                                      const float* __restrict__ y,
                                                      float* __restrict__ out)
{
    const int wave = threadIdx.x >> 5;
    const int lane = threadIdx.x & 31;
    const int b = blockIdx.x * 8 + wave;                // 0..1023
    float ie = 0.0f, cf = 0.0f;
#pragma unroll
    for (int s = 0; s < 2; ++s) {
        const int a   = lane + s * 32;                  // A = 64
        const int idx = b * 64 + a;
        const int eid = E[idx];                         // 0..3
        const int cid = C[idx] + 1;                     // 0..2
        if ((unsigned)eid < 4u) ie += y[(size_t)eid * NATOM + idx];
        if ((unsigned)cid < 3u) cf += y[(size_t)(4 + cid) * NATOM + idx];
    }
#pragma unroll
    for (int off = 16; off; off >>= 1) {
        ie += __shfl_xor(ie, off, 32);
        cf += __shfl_xor(cf, off, 32);
    }
    if (lane == 0) out[b] = -ie * cf;
}

// ---------------------------------------------------------------------------
// d_in order: [0]=X [1]=E [2]=C [3..13]=elem_Ws [14..24]=elem_bs
//             [25..35]=chg_Ws [36..46]=chg_bs
// ---------------------------------------------------------------------------
extern "C" void kernel_launch(void* const* d_in, const int* in_sizes, int n_in,
                              void* d_out, int out_size, void* d_ws, size_t ws_size,
                              hipStream_t stream)
{
    (void)in_sizes; (void)n_in; (void)out_size; (void)ws_size;
    const float* X = (const float*)d_in[0];
    const int*   E = (const int*)d_in[1];
    const int*   C = (const int*)d_in[2];

    WPtrs p;
    for (int l = 0; l < NLAYER; ++l) {
        p.W [l]          = (const float*)d_in[3  + l];   // elem_Ws
        p.Bv[l]          = (const float*)d_in[14 + l];   // elem_bs
        p.W [NLAYER + l] = (const float*)d_in[25 + l];   // chg_Ws
        p.Bv[NLAYER + l] = (const float*)d_in[36 + l];   // chg_bs
    }

    _Float16* wbuf = (_Float16*)d_ws;
    float*    bbuf = (float*)((char*)d_ws + WBUF_BYTES);
    float*    ybuf = (float*)((char*)d_ws + WBUF_BYTES + BBUF_BYTES);
    float*    out  = (float*)d_out;

    prep_kernel<<<NEXPERT * NLAYER, 256, 0, stream>>>(p, wbuf, bbuf);

    const int lds_bytes = TILE_M * 512 * 2 + NEXPERT * TILE_M * 128 * 2;   // 90,112
    funnel_kernel<<<NTILES, 224, lds_bytes, stream>>>(X, wbuf, bbuf, ybuf);

    combine_kernel<<<1024 / 8, 256, 0, stream>>>(E, C, ybuf, out);
}
